// CNN_TreeLSTM_37048387895819
// MI455X (gfx1250) — matrix operand order repaired and checked by reading, not compiled, not run on hardware
//
#include <hip/hip_runtime.h>
#include <hip/hip_bf16.h>

// ---------------- problem constants ----------------
#define Bdim   64
#define Ldim   40
#define Edim   300
#define EpadD  320   // E padded to mult of 32
#define Hdim   200
#define HpD    224   // H padded to mult of 32
#define KNUMD  200
#define KNpD   208   // filters padded to mult of 16
#define LrowsD 56    // padded sequence rows for conv input (covers l in [-2, 49])
#define LoutpD 48    // padded conv output length (3 N-tiles of 16)
#define NLABD  5

typedef _Float16 half_t;
typedef _Float16 v16h __attribute__((ext_vector_type(16)));
typedef _Float16 v8h  __attribute__((ext_vector_type(8)));
typedef float    v8f  __attribute__((ext_vector_type(8)));

// ---- WMMA operand loaders (contiguous k order, same for A and B) ----
// A 16x32: lanes0-15 hold M=lane, halves j -> K=j; lanes16-31 -> K=16+j. Contiguous.
__device__ __forceinline__ v16h load_a(const half_t* row, int kc, int kb) {
  v8h lo = *(const v8h*)(row + kc + 2 * kb);
  v8h hi = *(const v8h*)(row + kc + 2 * kb + 8);
  return __builtin_shufflevector(lo, hi, 0,1,2,3,4,5,6,7,8,9,10,11,12,13,14,15);
}
// B 32x16: lanes0-15 hold N=lane, halves j -> K=j; lanes16-31 -> K=16+j. Contiguous.
__device__ __forceinline__ v16h load_b(const half_t* col, int kc, int kb2) {
  return *(const v16h*)(col + kc + kb2);
}

__device__ __forceinline__ float sigmoidf_(float x) { return 1.0f / (1.0f + expf(-x)); }

// ---------------- prep kernels ----------------
__global__ void zero_h_kernel(half_t* p, int n) {
  int i = blockIdx.x * blockDim.x + threadIdx.x;
  if (i < n) p[i] = (half_t)0.0f;
}

__global__ void fh_init_kernel(float* FH, float* allC, float* allH, const float* bfh) {
  int i = blockIdx.x * blockDim.x + threadIdx.x;
  if (i < Bdim * Ldim * Hdim) {
    FH[i] = bfh[i % Hdim];  // all_H = 0  =>  FH = bfh
    allC[i] = 0.0f;
    allH[i] = 0.0f;
  }
}

// xe = emb[x]; thpad = tanh(xe) padded [B][Lrows][Epad]; xef = xe padded [B*L][Epad]
__global__ void embed_kernel(const int* x, const float* emb, half_t* thpad, half_t* xef) {
  int bl = blockIdx.x;           // 0..B*L-1
  int e  = threadIdx.x;          // 0..319
  int b = bl / Ldim, l = bl % Ldim;
  float v = 0.0f;
  if (e < Edim) v = emb[(size_t)x[bl] * Edim + e];
  thpad[((size_t)b * LrowsD + l + 2) * EpadD + e] = (half_t)tanhf(v);
  xef[(size_t)bl * EpadD + e] = (half_t)v;
}

// conv weights (KNUM,1,K,E) -> f16 padded [KNp][K][Epad]
__global__ void convw_kernel(const float* cw, int K, half_t* wpad) {
  int i = blockIdx.x * blockDim.x + threadIdx.x;
  int tot = KNpD * K * EpadD;
  if (i >= tot) return;
  int e = i % EpadD;
  int rest = i / EpadD;
  int kk = rest % K;
  int o  = rest / K;
  float v = (o < KNUMD && e < Edim) ? cw[((size_t)o * K + kk) * Edim + e] : 0.0f;
  wpad[i] = (half_t)v;
}

// W (fin x H) -> transposed f16 padded dst[ho][finPad]
__global__ void wT_kernel(const float* W, int fin, int finPad, half_t* dst) {
  int i = blockIdx.x * blockDim.x + threadIdx.x;
  int tot = 208 * finPad;
  if (i >= tot) return;
  int k  = i % finPad;
  int ho = i / finPad;
  float v = (ho < Hdim && k < fin) ? W[(size_t)k * Hdim + ho] : 0.0f;
  dst[i] = (half_t)v;
}

// ---------------- big GEMMs (WMMA) ----------------
// X = xef(2560 x 320) @ wT^T(320 x 200) + bias -> Xout(2560 x 200)
__global__ void x_gemm_kernel(const half_t* __restrict__ xef,
                              const half_t* __restrict__ wT,
                              const float* __restrict__ bias,
                              float* __restrict__ Xout) {
  int lane = threadIdx.x;
  int m0 = blockIdx.x * 16;
  int n0 = blockIdx.y * 16;
  int idx = lane & 15;
  int kb  = (lane < 16) ? 0 : 8;
  int kb2 = (lane < 16) ? 0 : 16;
  const half_t* arow = xef + (size_t)(m0 + idx) * EpadD;
  const half_t* bcol = wT  + (size_t)(n0 + idx) * EpadD;
  v8f c = {};
  for (int kc = 0; kc < EpadD; kc += 32) {
    v16h a  = load_a(arow, kc, kb);
    v16h bm = load_b(bcol, kc, kb2);
    c = __builtin_amdgcn_wmma_f32_16x16x32_f16(false, a, false, bm, (short)0, c, false, false);
  }
  int n = n0 + idx;
  if (n < Hdim) {
    int mrow = m0 + ((lane >> 4) << 3);
    float bv = bias[n];
#pragma unroll
    for (int r = 0; r < 8; ++r)
      Xout[(size_t)(mrow + r) * Hdim + n] = c[r] + bv;
  }
}

// conv GEMM: y[b][o][n] = tanh( sum_{kk,e} w[o][kk][e] * th[b][n+kk-pad][e] + cb[o] )
__global__ void conv_gemm_kernel(const half_t* __restrict__ thpad,
                                 const half_t* __restrict__ wpad,
                                 const float* __restrict__ cb,
                                 int K, float* __restrict__ y) {
  int lane = threadIdx.x;
  int b  = blockIdx.x;
  int m0 = blockIdx.y * 16;  // filter tile (13 tiles -> 208)
  int n0 = blockIdx.z * 16;  // position tile (3 tiles -> 48)
  int pad = K >> 1;
  int idx = lane & 15;
  int kb  = (lane < 16) ? 0 : 8;
  int kb2 = (lane < 16) ? 0 : 16;
  v8f c = {};
  for (int kk = 0; kk < K; ++kk) {
    const half_t* arow = wpad + ((size_t)(m0 + idx) * K + kk) * EpadD;
    int lrow = n0 + idx + kk - pad + 2;  // in [0, Lrows)
    const half_t* bcol = thpad + ((size_t)b * LrowsD + lrow) * EpadD;
    for (int kc = 0; kc < EpadD; kc += 32) {
      v16h a  = load_a(arow, kc, kb);
      v16h bm = load_b(bcol, kc, kb2);
      c = __builtin_amdgcn_wmma_f32_16x16x32_f16(false, a, false, bm, (short)0, c, false, false);
    }
  }
  int n = n0 + idx;
  int obase = m0 + ((lane >> 4) << 3);
#pragma unroll
  for (int r = 0; r < 8; ++r) {
    int o = obase + r;  // < 208
    float bias = (o < KNUMD) ? cb[o] : 0.0f;
    y[((size_t)b * KNpD + o) * LoutpD + n] = tanhf(c[r] + bias);
  }
}

// HS[z] = hsum16(64 x 224) @ whT[z]^T -> (64 x 200), no bias
__global__ void hs_gemm_kernel(const half_t* __restrict__ hsum16,
                               const half_t* __restrict__ whT,
                               float* __restrict__ HS) {
  int lane = threadIdx.x;
  int m0 = blockIdx.x * 16;  // 4 tiles -> 64
  int n0 = blockIdx.y * 16;  // 13 tiles -> 208
  int w  = blockIdx.z;       // 0:Wih 1:Wfh 2:Wuh
  int idx = lane & 15;
  int kb  = (lane < 16) ? 0 : 8;
  int kb2 = (lane < 16) ? 0 : 16;
  const half_t* arow = hsum16 + (size_t)(m0 + idx) * HpD;
  const half_t* bcol = whT + (size_t)w * 208 * HpD + (size_t)(n0 + idx) * HpD;
  v8f c = {};
  for (int kc = 0; kc < HpD; kc += 32) {
    v16h a  = load_a(arow, kc, kb);
    v16h bm = load_b(bcol, kc, kb2);
    c = __builtin_amdgcn_wmma_f32_16x16x32_f16(false, a, false, bm, (short)0, c, false, false);
  }
  int n = n0 + idx;
  if (n < Hdim) {
    int mrow = m0 + ((lane >> 4) << 3);
    float* out = HS + (size_t)w * Bdim * Hdim;
#pragma unroll
    for (int r = 0; r < 8; ++r)
      out[(size_t)(mrow + r) * Hdim + n] = c[r];
  }
}

// FH[b, node_b, :] = hnew16(64 x 224) @ WfhT + bfh
__global__ void fh_gemm_kernel(const half_t* __restrict__ hnew16,
                               const half_t* __restrict__ wfhT,
                               const float* __restrict__ bfh,
                               const int* __restrict__ bfs, int t,
                               float* __restrict__ FH) {
  int lane = threadIdx.x;
  int m0 = blockIdx.x * 16;
  int n0 = blockIdx.y * 16;
  int idx = lane & 15;
  int kb  = (lane < 16) ? 0 : 8;
  int kb2 = (lane < 16) ? 0 : 16;
  const half_t* arow = hnew16 + (size_t)(m0 + idx) * HpD;
  const half_t* bcol = wfhT + (size_t)(n0 + idx) * HpD;
  v8f c = {};
  for (int kc = 0; kc < HpD; kc += 32) {
    v16h a  = load_a(arow, kc, kb);
    v16h bm = load_b(bcol, kc, kb2);
    c = __builtin_amdgcn_wmma_f32_16x16x32_f16(false, a, false, bm, (short)0, c, false, false);
  }
  int n = n0 + idx;
  if (n < Hdim) {
    int mrow = m0 + ((lane >> 4) << 3);
    float bv = bfh[n];
#pragma unroll
    for (int r = 0; r < 8; ++r) {
      int bb = mrow + r;  // batch < 64
      int node = bfs[bb * Ldim + t];
      FH[((size_t)bb * Ldim + node) * Hdim + n] = c[r] + bv;
    }
  }
}

// ---------------- elementwise / reduction kernels ----------------
__global__ void convmax_kernel(const float* __restrict__ y, float* __restrict__ cnn_out) {
  int b = blockIdx.x;
  int j = threadIdx.x;
  if (j >= 3 * KNUMD) return;
  int ci = j / KNUMD;
  int o  = j % KNUMD;
  int lout = Ldim + ((ci == 1) ? 1 : 0);  // K=4 -> 41
  const float* row = y + ((size_t)ci * Bdim + b) * KNpD * LoutpD + (size_t)o * LoutpD;
  float m = -3.4e38f;
  for (int l = 0; l < lout; ++l) m = fmaxf(m, row[l]);
  cnn_out[b * 600 + j] = m;
}

__global__ void hsum_kernel(const int* __restrict__ bfs, const int* __restrict__ children,
                            const float* __restrict__ allH, int t,
                            float* __restrict__ hsum, half_t* __restrict__ hsum16) {
  int i = blockIdx.x * blockDim.x + threadIdx.x;
  if (i >= Bdim * Hdim) return;
  int b = i / Hdim, h = i % Hdim;
  int node = bfs[b * Ldim + t];
  const int* ch = children + ((size_t)b * Ldim + node) * Ldim;
  const float* Hb = allH + (size_t)b * Ldim * Hdim + h;
  float s = 0.0f;
  for (int l = 0; l < Ldim; ++l) s += (float)ch[l] * Hb[(size_t)l * Hdim];
  hsum[i] = s;
  hsum16[b * HpD + h] = (half_t)s;
}

__global__ void gates_kernel(const int* __restrict__ bfs, const int* __restrict__ children,
                             const float* __restrict__ X, const float* __restrict__ HS,
                             const float* __restrict__ FH,
                             const float* __restrict__ bih, const float* __restrict__ bfh,
                             const float* __restrict__ buh,
                             float* __restrict__ allC, float* __restrict__ allH,
                             half_t* __restrict__ hnew16, int t) {
  int i = blockIdx.x * blockDim.x + threadIdx.x;
  if (i >= Bdim * Hdim) return;
  const size_t BLH = (size_t)Bdim * Ldim * Hdim;
  const int BH = Bdim * Hdim;
  int b = i / Hdim, h = i % Hdim;
  int node = bfs[b * Ldim + t];
  size_t bn = ((size_t)b * Ldim + node) * Hdim + h;
  float fx = X[BLH + bn];  // XF (includes bfx)
  float ig = sigmoidf_(X[bn] + HS[i] + bih[h]);               // XI + hs@Wih + bih
  float og = sigmoidf_(fx + HS[BH + i] + bfh[h]);             // fx + hs@Wfh + bfh
  float ug = tanhf(X[2 * BLH + bn] + HS[2 * BH + i] + buh[h]); // XU + hs@Wuh + buh
  const int* ch = children + ((size_t)b * Ldim + node) * Ldim;
  float fc = 0.0f;
  for (int l = 0; l < Ldim; ++l) {
    size_t bl = ((size_t)b * Ldim + l) * Hdim + h;
    float fv = sigmoidf_(FH[bl] + fx);  // FH = all_H@Wfh + bfh (incremental)
    fc += (float)ch[l] * fv * allC[bl];
  }
  float cg = ig * ug + fc;
  float hn = og * tanhf(cg);
  allC[bn] = cg;
  allH[bn] = hn;
  hnew16[b * HpD + h] = (half_t)hn;
}

__global__ void treemax_kernel(const float* __restrict__ allH, float* __restrict__ tout) {
  int i = blockIdx.x * blockDim.x + threadIdx.x;
  if (i >= Bdim * Hdim) return;
  int b = i / Hdim, h = i % Hdim;
  float m = -3.4e38f;
  for (int l = 0; l < Ldim; ++l)
    m = fmaxf(m, tanhf(allH[((size_t)b * Ldim + l) * Hdim + h]));
  tout[i] = m;
}

__global__ void head_kernel(const float* __restrict__ cnn_out, const float* __restrict__ tout,
                            const float* __restrict__ W1, const float* __restrict__ b1,
                            const float* __restrict__ W2, const float* __restrict__ b2,
                            float* __restrict__ out) {
  __shared__ float feat[800];
  __shared__ float mid[200];
  int b = blockIdx.x;
  int tid = threadIdx.x;
  for (int j = tid; j < 800; j += 256) {
    float v = (j < 600) ? cnn_out[b * 600 + j] : tout[b * Hdim + (j - 600)];
    feat[j] = fmaxf(v, 0.0f);
  }
  __syncthreads();
  for (int j2 = tid; j2 < 200; j2 += 256) {
    float s = b1[j2];
    for (int j = 0; j < 800; ++j) s += feat[j] * W1[(size_t)j * 200 + j2];
    mid[j2] = fmaxf(s, 0.0f);  // relu(out1), only consumed through relu
  }
  __syncthreads();
  if (tid < NLABD) {
    float s = b2[tid];
    for (int j2 = 0; j2 < 200; ++j2) s += mid[j2] * W2[j2 * NLABD + tid];
    out[b * NLABD + tid] = s;
  }
}

// ---------------- host launcher ----------------
extern "C" void kernel_launch(void* const* d_in, const int* in_sizes, int n_in,
                              void* d_out, int out_size, void* d_ws, size_t ws_size,
                              hipStream_t stream) {
  const int*   x        = (const int*)d_in[0];
  const int*   bfs      = (const int*)d_in[1];
  const int*   children = (const int*)d_in[2];
  const float* emb      = (const float*)d_in[3];
  const float* cw0 = (const float*)d_in[4];
  const float* cb0 = (const float*)d_in[5];
  const float* cw1 = (const float*)d_in[6];
  const float* cb1 = (const float*)d_in[7];
  const float* cw2 = (const float*)d_in[8];
  const float* cb2 = (const float*)d_in[9];
  const float* Wix = (const float*)d_in[10];
  const float* bix = (const float*)d_in[11];
  const float* Wih = (const float*)d_in[12];
  const float* bih = (const float*)d_in[13];
  const float* Wfx = (const float*)d_in[14];
  const float* bfx = (const float*)d_in[15];
  const float* Wfh = (const float*)d_in[16];
  const float* bfh = (const float*)d_in[17];
  const float* Wux = (const float*)d_in[18];
  const float* bux = (const float*)d_in[19];
  const float* Wuh = (const float*)d_in[20];
  const float* buh = (const float*)d_in[21];
  const float* W1  = (const float*)d_in[22];
  const float* b1  = (const float*)d_in[23];
  const float* W2  = (const float*)d_in[24];
  const float* b2  = (const float*)d_in[25];
  float* out = (float*)d_out;

  // ---- carve workspace ----
  char* p = (char*)d_ws;
  auto carve = [&](size_t bytes) -> char* {
    char* r = p;
    p += (bytes + 255) & ~(size_t)255;
    return r;
  };
  half_t* thpad  = (half_t*)carve((size_t)Bdim * LrowsD * EpadD * 2);
  half_t* xef    = (half_t*)carve((size_t)Bdim * Ldim * EpadD * 2);
  half_t* wpad   = (half_t*)carve((size_t)KNpD * 12 * EpadD * 2);
  half_t* wxT    = (half_t*)carve((size_t)3 * 208 * EpadD * 2);
  half_t* whT    = (half_t*)carve((size_t)3 * 208 * HpD * 2);
  float*  X      = (float*)carve((size_t)3 * Bdim * Ldim * Hdim * 4);
  float*  Y      = (float*)carve((size_t)3 * Bdim * KNpD * LoutpD * 4);
  float*  cnnout = (float*)carve((size_t)Bdim * 600 * 4);
  float*  allC   = (float*)carve((size_t)Bdim * Ldim * Hdim * 4);
  float*  allH   = (float*)carve((size_t)Bdim * Ldim * Hdim * 4);
  float*  FH     = (float*)carve((size_t)Bdim * Ldim * Hdim * 4);
  float*  hsum   = (float*)carve((size_t)Bdim * Hdim * 4);
  half_t* hsum16 = (half_t*)carve((size_t)Bdim * HpD * 2);
  float*  HS     = (float*)carve((size_t)3 * Bdim * Hdim * 4);
  half_t* hnew16 = (half_t*)carve((size_t)Bdim * HpD * 2);
  float*  tout   = (float*)carve((size_t)Bdim * Hdim * 4);

  half_t* wpadK[3] = {wpad, wpad + (size_t)KNpD * 3 * EpadD, wpad + (size_t)KNpD * 7 * EpadD};
  const float* cwA[3] = {cw0, cw1, cw2};
  const float* cbA[3] = {cb0, cb1, cb2};
  const int KA[3] = {3, 4, 5};

  // ---- init / prep ----
  {
    int n = Bdim * LrowsD * EpadD;
    zero_h_kernel<<<(n + 255) / 256, 256, 0, stream>>>(thpad, n);
  }
  zero_h_kernel<<<(Bdim * HpD + 255) / 256, 256, 0, stream>>>(hsum16, Bdim * HpD);
  zero_h_kernel<<<(Bdim * HpD + 255) / 256, 256, 0, stream>>>(hnew16, Bdim * HpD);
  fh_init_kernel<<<(Bdim * Ldim * Hdim + 255) / 256, 256, 0, stream>>>(FH, allC, allH, bfh);

  embed_kernel<<<Bdim * Ldim, EpadD, 0, stream>>>(x, emb, thpad, xef);

  for (int i = 0; i < 3; ++i) {
    int tot = KNpD * KA[i] * EpadD;
    convw_kernel<<<(tot + 255) / 256, 256, 0, stream>>>(cwA[i], KA[i], wpadK[i]);
  }
  {
    int totE = 208 * EpadD, totH = 208 * HpD;
    wT_kernel<<<(totE + 255) / 256, 256, 0, stream>>>(Wix, Edim, EpadD, wxT);
    wT_kernel<<<(totE + 255) / 256, 256, 0, stream>>>(Wfx, Edim, EpadD, wxT + (size_t)208 * EpadD);
    wT_kernel<<<(totE + 255) / 256, 256, 0, stream>>>(Wux, Edim, EpadD, wxT + (size_t)2 * 208 * EpadD);
    wT_kernel<<<(totH + 255) / 256, 256, 0, stream>>>(Wih, Hdim, HpD, whT);
    wT_kernel<<<(totH + 255) / 256, 256, 0, stream>>>(Wfh, Hdim, HpD, whT + (size_t)208 * HpD);
    wT_kernel<<<(totH + 255) / 256, 256, 0, stream>>>(Wuh, Hdim, HpD, whT + (size_t)2 * 208 * HpD);
  }

  // ---- X precompute: XI/XF/XU = xe @ W + b (WMMA) ----
  const size_t BLH = (size_t)Bdim * Ldim * Hdim;
  const float* xbias[3] = {bix, bfx, bux};
  for (int w = 0; w < 3; ++w)
    x_gemm_kernel<<<dim3(Bdim * Ldim / 16, 13), 32, 0, stream>>>(
        xef, wxT + (size_t)w * 208 * EpadD, xbias[w], X + (size_t)w * BLH);

  // ---- convs (WMMA) + max pool ----
  for (int i = 0; i < 3; ++i)
    conv_gemm_kernel<<<dim3(Bdim, 13, 3), 32, 0, stream>>>(
        thpad, wpadK[i], cbA[i], KA[i], Y + (size_t)i * Bdim * KNpD * LoutpD);
  convmax_kernel<<<Bdim, 640, 0, stream>>>(Y, cnnout);

  // ---- Tree-LSTM recurrence ----
  const int bhBlocks = (Bdim * Hdim + 255) / 256;
  for (int t = 0; t < Ldim; ++t) {
    hsum_kernel<<<bhBlocks, 256, 0, stream>>>(bfs, children, allH, t, hsum, hsum16);
    hs_gemm_kernel<<<dim3(4, 13, 3), 32, 0, stream>>>(hsum16, whT, HS);
    gates_kernel<<<bhBlocks, 256, 0, stream>>>(bfs, children, X, HS, FH,
                                               bih, bfh, buh, allC, allH, hnew16, t);
    fh_gemm_kernel<<<dim3(4, 13), 32, 0, stream>>>(hnew16, whT + (size_t)208 * HpD,
                                                   bfh, bfs, t, FH);
  }

  // ---- head ----
  treemax_kernel<<<bhBlocks, 256, 0, stream>>>(allH, tout);
  head_kernel<<<Bdim, 256, 0, stream>>>(cnnout, tout, W1, b1, W2, b2, out);
}
